// Gemma4TextAttention_53601191854594
// MI455X (gfx1250) — matrix operand order repaired and checked by reading, not compiled, not run on hardware
//
#include <hip/hip_runtime.h>
#include <hip/hip_bf16.h>

#define S_LEN   4096
#define HID     2048
#define NH      8
#define NKV     4
#define HD      256
#define WINDOW  1024
#define SOFTCAP 50.0f

typedef __bf16 bf16;
typedef __attribute__((ext_vector_type(16))) __bf16 v16bf;
typedef __attribute__((ext_vector_type(8)))  float  v8f;
typedef __attribute__((ext_vector_type(4)))  unsigned u32x4;

union Frag { v16bf v; u32x4 q[2]; };

__device__ __forceinline__ bf16 f2bf(float f) {
    union { float f; unsigned u; } in; in.f = f;
    unsigned r = in.u + 0x7fffu + ((in.u >> 16) & 1u);
    union { unsigned short s; bf16 b; } out;
    out.s = (unsigned short)(r >> 16);
    return out.b;
}

// ---------------- TDM (Tensor Data Mover) support, guarded ----------------

#if defined(__HIP_DEVICE_COMPILE__) && __has_builtin(__builtin_amdgcn_tensor_load_to_lds) && \
    __has_builtin(__builtin_amdgcn_s_wait_tensorcnt)
#define HAVE_TDM 1
#else
#define HAVE_TDM 0
#endif

#if HAVE_TDM
typedef __attribute__((ext_vector_type(4))) unsigned tdm_u4;
typedef __attribute__((ext_vector_type(8))) int      tdm_i8;
typedef __attribute__((ext_vector_type(4))) int      tdm_i4;

// 2D tile load: nlines rows of line_elems bf16 elements, global line stride
// stride_elems, packed contiguously into LDS at lds_byte_addr.
// D# layout per CDNA5 ISA (group0: count/lds/global/type,
// group1: data_size=2B, tensor dims, tile dims, dim0 stride).
__device__ __forceinline__ void tdm_load_2d(const bf16* gsrc, unsigned lds_byte_addr,
                                            int line_elems, int nlines,
                                            int stride_elems, int td0, int td1) {
    unsigned long long ga = (unsigned long long)(size_t)gsrc;
    tdm_u4 g0;
    g0[0] = 1u;                                   // count=1, user descriptor
    g0[1] = lds_byte_addr;                        // lds_addr
    g0[2] = (unsigned)ga;                         // global_addr[31:0]
    g0[3] = (unsigned)((ga >> 32) & 0x01FFFFFFu)  // global_addr[56:32]
          | (2u << 30);                           // type=2 ("image")
    tdm_i8 g1;
    g1[0] = 0x00010000;                           // data_size=1 -> 2 bytes
    g1[1] = (td0 & 0xFFFF) << 16;                 // tensor_dim0[15:0]
    g1[2] = (int)(((unsigned)td0 >> 16) | ((unsigned)(td1 & 0xFFFF) << 16));
    g1[3] = (int)(((unsigned)td1 >> 16) | ((unsigned)line_elems << 16)); // tile_dim0
    g1[4] = nlines;                               // tile_dim1 (tile_dim2 = 0)
    g1[5] = stride_elems;                         // tensor_dim0_stride[31:0]
    g1[6] = 0;
    g1[7] = 0;
    tdm_i4 z4 = {0, 0, 0, 0};
#if __clang_major__ >= 23
    tdm_i8 z8 = {0, 0, 0, 0, 0, 0, 0, 0};
    __builtin_amdgcn_tensor_load_to_lds(g0, g1, z4, z4, z8, 0);
#else
    __builtin_amdgcn_tensor_load_to_lds(g0, g1, z4, z4, 0);
#endif
}
#endif

// ---------------- cast / transpose kernels ----------------

__global__ void cast_bf16_kernel(const float* __restrict__ src, bf16* __restrict__ dst, int n) {
    int i = blockIdx.x * blockDim.x + threadIdx.x;
    if (i < n) dst[i] = f2bf(src[i]);
}

// Wt[4096][2048]: row n = column n of [Wq | Wk | Wv]
__global__ void build_wqkvt_kernel(const float* __restrict__ Wq, const float* __restrict__ Wk,
                                   const float* __restrict__ Wv, bf16* __restrict__ Wt) {
    int i = blockIdx.x * blockDim.x + threadIdx.x;   // over 4096*2048
    int n = i >> 11;
    int k = i & 2047;
    float v;
    if (n < 2048)       v = Wq[(size_t)k * 2048 + n];
    else if (n < 3072)  v = Wk[(size_t)k * 1024 + (n - 2048)];
    else                v = Wv[(size_t)k * 1024 + (n - 3072)];
    Wt[i] = f2bf(v);
}

// Wt[C][R] = W[R][C]^T
__global__ void transpose_cast_kernel(const float* __restrict__ W, bf16* __restrict__ Wt,
                                      int R, int C) {
    int i = blockIdx.x * blockDim.x + threadIdx.x;
    if (i >= R * C) return;
    int c = i / R, r = i % R;
    Wt[i] = f2bf(W[(size_t)r * C + c]);
}

// ---------------- WMMA GEMM: C[M][N] f32 = A[M][K]bf16 @ Bt[N][K]bf16^T ----------------
// 256 threads = 8 waves; block tile 128(M) x 64(N); K-step 32.
// B tile (64 rows x 32 k = 4KB) staged into double-buffered LDS via TDM
// (fallback: cooperative b128 copy). A fragments loaded directly from global.

__global__ void __launch_bounds__(256) gemm_bf16_kernel(const bf16* __restrict__ A,
                                                        const bf16* __restrict__ Bt,
                                                        float* __restrict__ C,
                                                        int M, int N, int K) {
    __shared__ __align__(16) bf16 Blds[2][64 * 32];

    int tid  = threadIdx.x;
    int lane = tid & 31;
    int wave = tid >> 5;
    int half = lane >> 4;
    int lm   = lane & 15;
    int row0 = blockIdx.y * 128 + wave * 16;
    int col0 = blockIdx.x * 64;

    v8f acc[4] = {};
    const bf16* arow = A + (size_t)(row0 + lm) * K;

#if HAVE_TDM
    // stage first B tile
    if (wave == 0) {
        tdm_load_2d(Bt + (size_t)col0 * K, (unsigned)(size_t)&Blds[0][0],
                    32, 64, K, K, N);
        __builtin_amdgcn_s_wait_tensorcnt(0);
    }
    __syncthreads();
#else
    {
        int brow = tid >> 2, bchunk = tid & 3;
        const u32x4* src = (const u32x4*)(Bt + (size_t)(col0 + brow) * K + bchunk * 8);
        *(u32x4*)&Blds[0][brow * 32 + bchunk * 8] = *src;
    }
    __syncthreads();
#endif

    int cur = 0;
    for (int k0 = 0; k0 < K; k0 += 32) {
        int k1 = k0 + 32;
        // prefetch/stage next B tile into the other buffer
        if (k1 < K) {
#if HAVE_TDM
            if (wave == 0)
                tdm_load_2d(Bt + (size_t)col0 * K + k1, (unsigned)(size_t)&Blds[cur ^ 1][0],
                            32, 64, K, K, N);
#else
            int brow = tid >> 2, bchunk = tid & 3;
            const u32x4* src = (const u32x4*)(Bt + (size_t)(col0 + brow) * K + k1 + bchunk * 8);
            *(u32x4*)&Blds[cur ^ 1][brow * 32 + bchunk * 8] = *src;
#endif
        }

        __builtin_prefetch(arow + k0 + 512, 0, 0);
        Frag a;
        a.q[0] = *(const u32x4*)(arow + k0 + half * 8);
        a.q[1] = *(const u32x4*)(arow + k0 + 16 + half * 8);
#pragma unroll
        for (int t = 0; t < 4; ++t) {
            Frag b;
            const bf16* brow16 = &Blds[cur][(t * 16 + lm) * 32 + half * 8];
            b.q[0] = *(const u32x4*)(brow16);
            b.q[1] = *(const u32x4*)(brow16 + 16);
            acc[t] = __builtin_amdgcn_wmma_f32_16x16x32_bf16(
                false, a.v, false, b.v, (short)0, acc[t], false, false);
        }

#if HAVE_TDM
        if (k1 < K && wave == 0) __builtin_amdgcn_s_wait_tensorcnt(0);
#endif
        __syncthreads();
        cur ^= 1;
    }

#pragma unroll
    for (int t = 0; t < 4; ++t)
#pragma unroll
        for (int r = 0; r < 8; ++r)
            C[(size_t)(row0 + r + half * 8) * N + col0 + t * 16 + lm] = acc[t][r];
}

// ---------------- RMSNorm + RoPE epilogue ----------------
// QKV[s][0:2048]=Q(8 heads), [2048:3072]=K(4 heads), [3072:4096]=V(4 heads)
// Qb[h][s][d], Kb[kh][s][d] row-major (d contiguous); Vt[vh][d][s] (keys contiguous)

__global__ void qkv_epilogue_kernel(const float* __restrict__ QKV,
                                    const float* __restrict__ q_scale,
                                    const float* __restrict__ k_scale,
                                    const int* __restrict__ positions,
                                    bf16* __restrict__ Qb, bf16* __restrict__ Kb,
                                    bf16* __restrict__ Vt) {
    __shared__ float red[256];
    __shared__ float buf[256];
    int s = blockIdx.x, tid = threadIdx.x;
    float pos = (float)positions[s];

    for (int seg = 0; seg < 16; ++seg) {
        float x = QKV[(size_t)s * 4096 + seg * 256 + tid];
        red[tid] = x * x;
        __syncthreads();
        for (int st = 128; st > 0; st >>= 1) {
            if (tid < st) red[tid] += red[tid + st];
            __syncthreads();
        }
        float rms = rsqrtf(red[0] * (1.0f / 256.0f) + 1e-6f);
        __syncthreads();

        if (seg < 12) {
            float w = (seg < 8) ? q_scale[tid] : k_scale[tid];
            float y = x * rms * (1.0f + w);
            buf[tid] = y;
            __syncthreads();
            if (tid < 128) {
                float x1 = buf[tid], x2 = buf[tid + 128];
                float freq = __powf(10000.0f, -(float)tid * (1.0f / 128.0f));
                float ang = pos * freq;
                float sn, cs;
                __sincosf(ang, &sn, &cs);
                float o1 = x1 * cs - x2 * sn;
                float o2 = x2 * cs + x1 * sn;
                if (seg < 8) {
                    bf16* dst = Qb + ((size_t)seg * S_LEN + s) * HD;
                    dst[tid] = f2bf(o1);
                    dst[tid + 128] = f2bf(o2);
                } else {
                    bf16* dst = Kb + ((size_t)(seg - 8) * S_LEN + s) * HD;
                    dst[tid] = f2bf(o1);
                    dst[tid + 128] = f2bf(o2);
                }
            }
            __syncthreads();
        } else {
            int vh = seg - 12;
            Vt[((size_t)vh * HD + tid) * S_LEN + s] = f2bf(x * rms);
            __syncthreads();
        }
    }
}

// ---------------- flash-style sliding-window attention ----------------
// one wave per (head, 16-query tile)

__global__ void __launch_bounds__(32) attn_kernel(const bf16* __restrict__ Qb,
                                                  const bf16* __restrict__ Kb,
                                                  const bf16* __restrict__ Vt,
                                                  bf16* __restrict__ AttnB) {
    __shared__ __align__(16) bf16 Plds[16 * 32];
    int lane = threadIdx.x & 31;
    int half = lane >> 4;
    int lm   = lane & 15;
    int h    = blockIdx.y;
    int q0   = blockIdx.x * 16;
    int kvh  = h >> 1;   // GROUPS = 2

    // Q tile: 16 x 256 as 8 A-fragments
    Frag qa[8];
    const bf16* qrow = Qb + ((size_t)h * S_LEN + q0 + lm) * HD;
#pragma unroll
    for (int i = 0; i < 8; ++i) {
        qa[i].q[0] = *(const u32x4*)(qrow + i * 32 + half * 8);
        qa[i].q[1] = *(const u32x4*)(qrow + i * 32 + 16 + half * 8);
    }

    v8f acc[16] = {};
    float m_r[8], l_r[8];
#pragma unroll
    for (int r = 0; r < 8; ++r) { m_r[r] = -3.0e38f; l_r[r] = 0.0f; }

    int kstart = q0 - (WINDOW - 1);
    if (kstart < 0) kstart = 0;
    kstart &= ~31;

    for (int kb = kstart; kb <= q0 + 15; kb += 32) {
        float p2[2][8];
        // logits for two 16-key tiles
#pragma unroll
        for (int tt = 0; tt < 2; ++tt) {
            v8f lg = {};
            int key  = kb + tt * 16 + lm;
            int keyc = key < (S_LEN - 1) ? key : (S_LEN - 1);
            const bf16* krow = Kb + ((size_t)kvh * S_LEN + keyc) * HD;
#pragma unroll
            for (int i = 0; i < 8; ++i) {
                Frag b;
                const u32x4* pb = (const u32x4*)(krow + i * 32 + half * 8);
                b.q[0] = pb[0];
                b.q[1] = pb[2];
                lg = __builtin_amdgcn_wmma_f32_16x16x32_bf16(
                    false, qa[i].v, false, b.v, (short)0, lg, false, false);
            }
#pragma unroll
            for (int r = 0; r < 8; ++r) {
                int q = q0 + r + half * 8;
                float x = tanhf(lg[r] * (1.0f / SOFTCAP)) * SOFTCAP;
                bool valid = (key <= q) && (q - key < WINDOW);
                p2[tt][r] = valid ? x : -3.0e38f;
            }
        }
        // online softmax
        float scale[8];
#pragma unroll
        for (int r = 0; r < 8; ++r) {
            float t = fmaxf(p2[0][r], p2[1][r]);
#pragma unroll
            for (int off = 1; off < 16; off <<= 1) t = fmaxf(t, __shfl_xor(t, off, 32));
            float mnew = fmaxf(m_r[r], t);
            float e0 = (p2[0][r] > -1.0e37f) ? __expf(p2[0][r] - mnew) : 0.0f;
            float e1 = (p2[1][r] > -1.0e37f) ? __expf(p2[1][r] - mnew) : 0.0f;
            float sm = e0 + e1;
#pragma unroll
            for (int off = 1; off < 16; off <<= 1) sm += __shfl_xor(sm, off, 32);
            scale[r] = __expf(m_r[r] - mnew);
            l_r[r] = l_r[r] * scale[r] + sm;
            m_r[r] = mnew;
            p2[0][r] = e0;
            p2[1][r] = e1;
        }
#pragma unroll
        for (int dt = 0; dt < 16; ++dt)
#pragma unroll
            for (int r = 0; r < 8; ++r) acc[dt][r] = acc[dt][r] * scale[r];

        // P (C-layout) -> LDS [row][key]
#pragma unroll
        for (int tt = 0; tt < 2; ++tt)
#pragma unroll
            for (int r = 0; r < 8; ++r)
                Plds[(r + half * 8) * 32 + tt * 16 + lm] = f2bf(p2[tt][r]);
        __syncthreads();

        // reload P as A fragment (16x32)
        Frag pa;
        {
            const bf16* prow = Plds + lm * 32;
            pa.q[0] = *(const u32x4*)(prow + half * 8);
            pa.q[1] = *(const u32x4*)(prow + 16 + half * 8);
        }

        // P @ V : 16 d-tiles
        const bf16* vbase = Vt + (size_t)kvh * HD * S_LEN;
#pragma unroll
        for (int dt = 0; dt < 16; ++dt) {
            Frag b;
            const u32x4* pb = (const u32x4*)(vbase + (size_t)(dt * 16 + lm) * S_LEN + kb + half * 8);
            b.q[0] = pb[0];
            b.q[1] = pb[2];
            acc[dt] = __builtin_amdgcn_wmma_f32_16x16x32_bf16(
                false, pa.v, false, b.v, (short)0, acc[dt], false, false);
        }
        __syncthreads();
    }

    // normalize + store bf16 for the output projection
#pragma unroll
    for (int dt = 0; dt < 16; ++dt)
#pragma unroll
        for (int r = 0; r < 8; ++r) {
            int q = q0 + r + half * 8;
            float o = acc[dt][r] / l_r[r];
            AttnB[(size_t)q * (NH * HD) + h * HD + dt * 16 + lm] = f2bf(o);
        }
}

// ---------------- launch ----------------

extern "C" void kernel_launch(void* const* d_in, const int* in_sizes, int n_in,
                              void* d_out, int out_size, void* d_ws, size_t ws_size,
                              hipStream_t stream) {
    const float* H        = (const float*)d_in[0];
    const float* Wq       = (const float*)d_in[1];
    const float* Wk       = (const float*)d_in[2];
    const float* Wv       = (const float*)d_in[3];
    const float* Wo       = (const float*)d_in[4];
    const float* q_scale  = (const float*)d_in[5];
    const float* k_scale  = (const float*)d_in[6];
    const int*   positions = (const int*)d_in[7];
    float* out = (float*)d_out;

    char* ws = (char*)d_ws;
    size_t off = 0;
    auto alloc = [&](size_t bytes) -> void* {
        void* p = ws + off;
        off = (off + bytes + 255) & ~(size_t)255;
        return p;
    };

    bf16*  Hb    = (bf16*)alloc((size_t)S_LEN * HID * 2);
    bf16*  Wqkvt = (bf16*)alloc((size_t)4096 * HID * 2);
    bf16*  Wot   = (bf16*)alloc((size_t)HID * HID * 2);
    float* QKV   = (float*)alloc((size_t)S_LEN * 4096 * 4);
    bf16*  Qb    = (bf16*)alloc((size_t)NH * S_LEN * HD * 2);
    bf16*  Kb    = (bf16*)alloc((size_t)NKV * S_LEN * HD * 2);
    bf16*  Vt    = (bf16*)alloc((size_t)NKV * HD * S_LEN * 2);
    bf16*  AttnB = (bf16*)alloc((size_t)S_LEN * NH * HD * 2);

    cast_bf16_kernel<<<(S_LEN * HID) / 256, 256, 0, stream>>>(H, Hb, S_LEN * HID);
    build_wqkvt_kernel<<<(4096 * HID) / 256, 256, 0, stream>>>(Wq, Wk, Wv, Wqkvt);
    transpose_cast_kernel<<<(HID * HID) / 256, 256, 0, stream>>>(Wo, Wot, HID, HID);

    gemm_bf16_kernel<<<dim3(4096 / 64, S_LEN / 128), 256, 0, stream>>>(Hb, Wqkvt, QKV,
                                                                       S_LEN, 4096, HID);

    qkv_epilogue_kernel<<<S_LEN, 256, 0, stream>>>(QKV, q_scale, k_scale, positions, Qb, Kb, Vt);

    attn_kernel<<<dim3(S_LEN / 16, NH), 32, 0, stream>>>(Qb, Kb, Vt, AttnB);

    gemm_bf16_kernel<<<dim3(HID / 64, S_LEN / 128), 256, 0, stream>>>(AttnB, Wot, out,
                                                                      S_LEN, HID, HID);
}
